// LSTM_FEAT_CLASS_24524263260277
// MI455X (gfx1250) — matrix operand order repaired and checked by reading, not compiled, not run on hardware
//
#include <hip/hip_runtime.h>
#include <hip/hip_bf16.h>
#include <stdint.h>

// ---------------- problem constants ----------------
#define T_STEPS 8192
#define IN_DIM  118
#define EMBED   64
#define HIDDEN  1024
#define G4      4096           // 4*HIDDEN gate rows
#define OUT_N   63
#define K0TOT   1216           // 192 (padded x||embed) + 1024 (h0)
#define NCH0    38             // K chunks of 32 for layer0
#define NCH1    64             // layer1: [h0;h1] = 2048
#define NCHH    32             // head:   h1 = 1024
#define NWG     32
#define TPB     256            // 8 wave32 per WG -> 256 waves = 256 row tiles

typedef __attribute__((ext_vector_type(16))) __bf16 v16bf;
typedef __attribute__((ext_vector_type(8)))  float  v8f;

// ---------------- workspace layout (bytes) ----------------
#define OFF_W0  0ull
#define SZ_W0   ((size_t)G4 * K0TOT * 2)
#define OFF_W1  (OFF_W0 + SZ_W0)
#define SZ_W1   ((size_t)G4 * 2048 * 2)
#define OFF_WH  (OFF_W1 + SZ_W1)
#define SZ_WH   ((size_t)64 * 1024 * 2)
#define OFF_B0  (OFF_WH + SZ_WH)
#define OFF_B1  (OFF_B0 + (size_t)G4 * 4)
#define OFF_BH  (OFF_B1 + (size_t)G4 * 4)
#define OFF_G0  (OFF_BH + 64 * 4)          // gates0[2][4096] f32 (parity buffered)
#define OFF_G1  (OFF_G0 + 2 * G4 * 4)
#define OFF_BAR (OFF_G1 + 2 * G4 * 4)      // 2 x u32 barrier state

__device__ __forceinline__ unsigned short f2bf(float f) {
    unsigned u = __float_as_uint(f);
    unsigned r = u + 0x7FFFu + ((u >> 16) & 1u);   // round-to-nearest-even
    return (unsigned short)(r >> 16);
}
__device__ __forceinline__ float sigm(float x) { return 1.f / (1.f + __expf(-x)); }

// A-matrix (16x32 bf16) lane/VGPR swizzle used by this code (contiguous K order, matched by the B loads):
// lanes 0-15 hold row M=lane, K = {0..15}; lanes 16-31 hold K = {16..31}.
__device__ __forceinline__ size_t amap(int row, int k, int nch) {
    int tm = row >> 4, rlo = row & 15, tk = k >> 5, klo = k & 31;
    int lane = rlo + ((klo & 16) ? 16 : 0);
    int pos  = klo & 15;
    return ((size_t)(tm * nch + tk) << 9) + ((size_t)lane << 4) + (size_t)pos;
}

// ---------------- prep: cast+swizzle weights to bf16 WMMA A-layout ----------------
__global__ void pack_w0(const float* __restrict__ wih, const float* __restrict__ whh,
                        unsigned short* __restrict__ dst) {
    int e = blockIdx.x * blockDim.x + threadIdx.x;
    if (e >= G4 * K0TOT) return;
    int row = e / K0TOT, k = e - row * K0TOT;
    float v;
    if (k < 182)      v = wih[row * 182 + k];          // x_t (118) || embed (64)
    else if (k < 192) v = 0.f;                         // pad
    else              v = whh[row * HIDDEN + (k - 192)];
    dst[amap(row, k, NCH0)] = f2bf(v);
}
__global__ void pack_w1(const float* __restrict__ wih, const float* __restrict__ whh,
                        unsigned short* __restrict__ dst) {
    int e = blockIdx.x * blockDim.x + threadIdx.x;
    if (e >= G4 * 2048) return;
    int row = e >> 11, k = e & 2047;
    float v = (k < 1024) ? wih[row * HIDDEN + k] : whh[row * HIDDEN + (k - 1024)];
    dst[amap(row, k, NCH1)] = f2bf(v);
}
__global__ void pack_wh(const float* __restrict__ wlin, unsigned short* __restrict__ dst) {
    int e = blockIdx.x * blockDim.x + threadIdx.x;
    if (e >= 64 * 1024) return;
    int row = e >> 10, k = e & 1023;
    float v = (row < OUT_N) ? wlin[row * HIDDEN + k] : 0.f;
    dst[amap(row, k, NCHH)] = f2bf(v);
}
__global__ void init_misc(const float* bih0, const float* bhh0,
                          const float* bih1, const float* bhh1,
                          const float* blin,
                          float* b0, float* b1, float* bh, unsigned* bar) {
    int i = blockIdx.x * blockDim.x + threadIdx.x;
    if (i < G4)              b0[i] = bih0[i] + bhh0[i];
    else if (i < 2 * G4)     { int j = i - G4; b1[j] = bih1[j] + bhh1[j]; }
    else if (i < 2 * G4 + 64){ int j = i - 2 * G4; bh[j] = (j < OUT_N) ? blin[j] : 0.f; }
    else if (i < 2 * G4 + 66) bar[i - 2 * G4 - 64] = 0u;
}

// ---------------- device-wide barrier (persistent kernel, 32 co-resident WGs) ----------------
__device__ __forceinline__ void grid_barrier(unsigned* cnt, unsigned* gen) {
    __syncthreads();
    if (threadIdx.x == 0) {
        __threadfence();
        unsigned g = __hip_atomic_load(gen, __ATOMIC_ACQUIRE, __HIP_MEMORY_SCOPE_AGENT);
        unsigned a = __hip_atomic_fetch_add(cnt, 1u, __ATOMIC_ACQ_REL, __HIP_MEMORY_SCOPE_AGENT);
        if (a == NWG - 1u) {
            __hip_atomic_store(cnt, 0u, __ATOMIC_RELAXED, __HIP_MEMORY_SCOPE_AGENT);
            __hip_atomic_fetch_add(gen, 1u, __ATOMIC_ACQ_REL, __HIP_MEMORY_SCOPE_AGENT);
        } else {
            while (__hip_atomic_load(gen, __ATOMIC_ACQUIRE, __HIP_MEMORY_SCOPE_AGENT) == g)
                __builtin_amdgcn_s_sleep(2);
        }
        __threadfence();
    }
    __syncthreads();
}

// ---------------- persistent LSTM kernel ----------------
__global__ void __launch_bounds__(TPB) lstm_main(
    const float* __restrict__ input,           // [8192,118]
    const float* __restrict__ wmap,            // [64,63] f32 (tiny, VALU)
    const float* __restrict__ bmap,            // [64]
    const unsigned short* __restrict__ w0,
    const unsigned short* __restrict__ w1,
    const unsigned short* __restrict__ wh,
    const float* __restrict__ b0, const float* __restrict__ b1, const float* __restrict__ bh,
    float* __restrict__ g0buf, float* __restrict__ g1buf,
    unsigned* __restrict__ bar,
    float* __restrict__ out)
{
    __shared__ __align__(32) unsigned short xbf[192];   // bf16 [x_t | embed | pad]
    __shared__ __align__(32) unsigned short h0b[HIDDEN];
    __shared__ __align__(32) unsigned short h1b[HIDDEN];
    __shared__ float c0s[HIDDEN], c1s[HIDDEN];
    __shared__ float ldsY[64], ldsY2[64];

    const int tid = threadIdx.x, bid = blockIdx.x;
    const int wid = tid >> 5, lane = tid & 31;
    const int gw = bid * 8 + wid;              // global wave = gate row-tile (0..255)
    const int laneSel = (lane >> 4) << 4;      // B operand: lanes 16-31 take K+16..K+31

    for (int i = tid; i < HIDDEN; i += TPB) { c0s[i] = 0.f; c1s[i] = 0.f; h0b[i] = 0; h1b[i] = 0; }
    for (int i = tid; i < 192; i += TPB) xbf[i] = (i < IN_DIM) ? f2bf(input[i]) : (unsigned short)0;
    __syncthreads();

    const v16bf* A0 = reinterpret_cast<const v16bf*>(w0);
    const v16bf* A1 = reinterpret_cast<const v16bf*>(w1);
    const v16bf* AH = reinterpret_cast<const v16bf*>(wh);

    #pragma unroll 1
    for (int t = 0; t < T_STEPS; ++t) {
        const int p = t & 1;

        // ---- gates0 = W0 . [x|embed|pad|h0], one 16-row tile per wave ----
        {
            v8f acc = {};
            const v16bf* Ab = A0 + (size_t)gw * (NCH0 * 32) + lane;
            #pragma unroll 2
            for (int c = 0; c < 6; ++c) {
                v16bf bv = *reinterpret_cast<const v16bf*>(xbf + c * 32 + laneSel);
                acc = __builtin_amdgcn_wmma_f32_16x16x32_bf16(false, Ab[c * 32], false, bv,
                                                              (short)0, acc, false, false);
            }
            #pragma unroll 2
            for (int c = 0; c < 32; ++c) {
                v16bf bv = *reinterpret_cast<const v16bf*>(h0b + c * 32 + laneSel);
                acc = __builtin_amdgcn_wmma_f32_16x16x32_bf16(false, Ab[(6 + c) * 32], false, bv,
                                                              (short)0, acc, false, false);
            }
            if ((lane & 15) == 0) {            // column 0 holds the GEMV result
                float* g = g0buf + p * G4 + gw * 16 + ((lane >> 4) << 3);
                #pragma unroll
                for (int v = 0; v < 8; ++v) g[v] = acc[v];
            }
        }
        grid_barrier(bar, bar + 1);

        // ---- h0/c0 update (redundant per WG; state lives in LDS) ----
        {
            const float* g = g0buf + p * G4;
            #pragma unroll
            for (int r = 0; r < 4; ++r) {
                int j = r * TPB + tid;
                float gi = g[j]              + b0[j];
                float gf = g[HIDDEN + j]     + b0[HIDDEN + j];
                float gg = g[2 * HIDDEN + j] + b0[2 * HIDDEN + j];
                float go = g[3 * HIDDEN + j] + b0[3 * HIDDEN + j];
                float c  = sigm(gf) * c0s[j] + sigm(gi) * tanhf(gg);
                c0s[j] = c;
                h0b[j] = f2bf(sigm(go) * tanhf(c));
            }
        }
        __syncthreads();

        // ---- gates1 = W1 . [h0;h1] ----
        {
            v8f acc = {};
            const v16bf* Ab = A1 + (size_t)gw * (NCH1 * 32) + lane;
            #pragma unroll 2
            for (int c = 0; c < 32; ++c) {
                v16bf bv = *reinterpret_cast<const v16bf*>(h0b + c * 32 + laneSel);
                acc = __builtin_amdgcn_wmma_f32_16x16x32_bf16(false, Ab[c * 32], false, bv,
                                                              (short)0, acc, false, false);
            }
            #pragma unroll 2
            for (int c = 0; c < 32; ++c) {
                v16bf bv = *reinterpret_cast<const v16bf*>(h1b + c * 32 + laneSel);
                acc = __builtin_amdgcn_wmma_f32_16x16x32_bf16(false, Ab[(32 + c) * 32], false, bv,
                                                              (short)0, acc, false, false);
            }
            if ((lane & 15) == 0) {
                float* g = g1buf + p * G4 + gw * 16 + ((lane >> 4) << 3);
                #pragma unroll
                for (int v = 0; v < 8; ++v) g[v] = acc[v];
            }
        }
        grid_barrier(bar, bar + 1);

        // ---- h1/c1 update (redundant per WG) ----
        {
            const float* g = g1buf + p * G4;
            #pragma unroll
            for (int r = 0; r < 4; ++r) {
                int j = r * TPB + tid;
                float gi = g[j]              + b1[j];
                float gf = g[HIDDEN + j]     + b1[HIDDEN + j];
                float gg = g[2 * HIDDEN + j] + b1[2 * HIDDEN + j];
                float go = g[3 * HIDDEN + j] + b1[3 * HIDDEN + j];
                float c  = sigm(gf) * c1s[j] + sigm(gi) * tanhf(gg);
                c1s[j] = c;
                h1b[j] = f2bf(sigm(go) * tanhf(c));
            }
        }
        __syncthreads();

        // ---- head y = W_lin . h1 (redundant per WG; waves 0..3, 4 row-tiles) ----
        if (wid < 4) {
            v8f acc = {};
            const v16bf* Ab = AH + (size_t)wid * (NCHH * 32) + lane;
            #pragma unroll 2
            for (int c = 0; c < NCHH; ++c) {
                v16bf bv = *reinterpret_cast<const v16bf*>(h1b + c * 32 + laneSel);
                acc = __builtin_amdgcn_wmma_f32_16x16x32_bf16(false, Ab[c * 32], false, bv,
                                                              (short)0, acc, false, false);
            }
            if ((lane & 15) == 0) {
                #pragma unroll
                for (int v = 0; v < 8; ++v)
                    ldsY[wid * 16 + ((lane >> 4) << 3) + v] = acc[v];
            }
        }
        __syncthreads();
        if (tid < 64) {
            float yb = ldsY[tid] + bh[tid];
            ldsY2[tid] = yb;
            if (bid == 0 && t == T_STEPS - 1 && tid < OUT_N) out[tid] = yb;  // final y
        }
        __syncthreads();
        // ---- embed = W_map . y + b_map; rebuild bf16 x-vector for step t+1 ----
        if (tid < 64) {
            float e = bmap[tid];
            for (int o = 0; o < OUT_N; ++o) e += wmap[tid * OUT_N + o] * ldsY2[o];
            xbf[IN_DIM + tid] = f2bf(e);
        }
        if (tid < IN_DIM && t + 1 < T_STEPS)
            xbf[tid] = f2bf(input[(size_t)(t + 1) * IN_DIM + tid]);
        __syncthreads();
    }
}

// ---------------- launch ----------------
extern "C" void kernel_launch(void* const* d_in, const int* in_sizes, int n_in,
                              void* d_out, int out_size, void* d_ws, size_t ws_size,
                              hipStream_t stream) {
    (void)in_sizes; (void)n_in; (void)out_size; (void)ws_size;
    const float* input = (const float*)d_in[0];
    const float* wih0  = (const float*)d_in[1];
    const float* whh0  = (const float*)d_in[2];
    const float* bih0  = (const float*)d_in[3];
    const float* bhh0  = (const float*)d_in[4];
    const float* wih1  = (const float*)d_in[5];
    const float* whh1  = (const float*)d_in[6];
    const float* bih1  = (const float*)d_in[7];
    const float* bhh1  = (const float*)d_in[8];
    const float* wlin  = (const float*)d_in[9];
    const float* blin  = (const float*)d_in[10];
    const float* wmap  = (const float*)d_in[11];
    const float* bmap  = (const float*)d_in[12];

    char* ws = (char*)d_ws;
    unsigned short* w0p = (unsigned short*)(ws + OFF_W0);
    unsigned short* w1p = (unsigned short*)(ws + OFF_W1);
    unsigned short* whp = (unsigned short*)(ws + OFF_WH);
    float* b0 = (float*)(ws + OFF_B0);
    float* b1 = (float*)(ws + OFF_B1);
    float* bh = (float*)(ws + OFF_BH);
    float* g0 = (float*)(ws + OFF_G0);
    float* g1 = (float*)(ws + OFF_G1);
    unsigned* bar = (unsigned*)(ws + OFF_BAR);

    pack_w0<<<(G4 * K0TOT + 255) / 256, 256, 0, stream>>>(wih0, whh0, w0p);
    pack_w1<<<(G4 * 2048 + 255) / 256, 256, 0, stream>>>(wih1, whh1, w1p);
    pack_wh<<<(64 * 1024 + 255) / 256, 256, 0, stream>>>(wlin, whp);
    init_misc<<<(2 * G4 + 66 + 255) / 256, 256, 0, stream>>>(bih0, bhh0, bih1, bhh1, blin,
                                                             b0, b1, bh, bar);
    lstm_main<<<NWG, TPB, 0, stream>>>(input, wmap, bmap, w0p, w1p, whp,
                                       b0, b1, bh, g0, g1, bar, (float*)d_out);
}
